// GCN_4587025072291
// MI455X (gfx1250) — matrix operand order repaired and checked by reading, not compiled, not run on hardware
//
#include <hip/hip_runtime.h>
#include <hip/hip_bf16.h>

// ---- CDNA5 WMMA vector types (wave32) ----
typedef __attribute__((ext_vector_type(16))) __bf16 v16bf;
typedef __attribute__((ext_vector_type(8)))  float  v8f;

#define NEG_SLOPE 0.01f

// ---------------- degree / normalization ----------------
__global__ void k_deg_init(float* __restrict__ deg, int N) {
    int i = blockIdx.x * blockDim.x + threadIdx.x;
    if (i < N) deg[i] = 1.0f;   // self-loop
}

__global__ void k_deg_count(const long long* __restrict__ dst,
                            float* __restrict__ deg, int E) {
    int e = blockIdx.x * blockDim.x + threadIdx.x;
    if (e < E) atomicAdd(&deg[(int)dst[e]], 1.0f);  // exact: integer-valued sums
}

__global__ void k_rsqrt_inplace(float* __restrict__ deg, int N) {
    int i = blockIdx.x * blockDim.x + threadIdx.x;
    if (i < N) deg[i] = rsqrtf(deg[i]);             // deg >= 1 always
}

// ---------------- WMMA GEMM: out[N x 64] = act(A[N x KDIM]) @ W[KDIM x 64] ----
// Block = 256 threads = 8 waves; each wave computes one 16x16 tile of a 32x64
// block tile. W is staged in LDS as bf16 *transposed* (sWT[col][K]) so each
// lane's B fragment (16 consecutive K for a fixed column) is one contiguous
// 32-byte LDS read (2x ds_load_b128) instead of 16 scalar ds_load_u16.
// A is converted fp32->bf16 on the fly (optionally through leaky-ReLU);
// accumulation is f32 via v_wmma_f32_16x16x32_bf16.
template <int KDIM, bool LRELU>
__global__ __launch_bounds__(256) void k_gemm_wmma(
    const float* __restrict__ A, const float* __restrict__ W,
    float* __restrict__ out, int N) {
    __shared__ __bf16 sWT[64 * KDIM];   // [col][K], bf16

    // cooperative transpose-load of W (KDIM x 64, row-major) into sWT
    for (int i = threadIdx.x; i < 64 * KDIM; i += 256) {
        int col = i / KDIM;
        int k   = i - col * KDIM;
        sWT[i] = (__bf16)W[k * 64 + col];
    }
    __syncthreads();

    const int lane = threadIdx.x & 31;
    const int wave = threadIdx.x >> 5;
    const int wrow = wave >> 2;        // 0..1  -> which 16-row strip
    const int wcol = wave & 3;         // 0..3  -> which 16-col strip
    const int half = lane >> 4;        // 0..1
    const int l16  = lane & 15;

    const int rowA   = blockIdx.x * 32 + wrow * 16 + l16;   // A-frag row (M)
    const int rclamp = rowA < N ? rowA : (N - 1);           // clamp: keep EXEC full
    const float* ap  = A + (size_t)rclamp * KDIM;
    const int colB   = wcol * 16 + l16;                     // B-frag col (N)
    const __bf16* wp = &sWT[colB * KDIM + 16 * half];       // contiguous K run

    v8f acc = {};
#pragma unroll
    for (int kk = 0; kk < KDIM; kk += 32) {
        // A 16x32 bf16 layout: element e -> K = 16*half + e
        // -> one contiguous 16-float run per lane: [kk+16h, +16)
        float4 r0 = *(const float4*)(ap + kk + 16 * half);
        float4 r1 = *(const float4*)(ap + kk + 16 * half + 4);
        float4 r2 = *(const float4*)(ap + kk + 16 * half + 8);
        float4 r3 = *(const float4*)(ap + kk + 16 * half + 12);
        float f[16] = {r0.x, r0.y, r0.z, r0.w, r1.x, r1.y, r1.z, r1.w,
                       r2.x, r2.y, r2.z, r2.w, r3.x, r3.y, r3.z, r3.w};
        v16bf a;
#pragma unroll
        for (int e = 0; e < 16; ++e) {
            float v = f[e];
            if (LRELU) v = v > 0.0f ? v : NEG_SLOPE * v;
            a[e] = (__bf16)v;
        }
        // B 32x16 bf16 layout: element e -> K = kk + 16*half + e, N = lane&15
        // contiguous 16 bf16 (32 B, 32-B aligned) in transposed LDS
        v16bf b = *(const v16bf*)(wp + kk);

        acc = __builtin_amdgcn_wmma_f32_16x16x32_bf16(
            /*neg_a=*/false, a, /*neg_b=*/false, b,
            /*c_mod=*/(short)0, acc, /*reuse_a=*/false, /*reuse_b=*/false);
    }

    // D 16x16 f32 layout: VGPR r -> M = r + 8*half, N = lane&15
#pragma unroll
    for (int r = 0; r < 8; ++r) {
        int m = blockIdx.x * 32 + wrow * 16 + r + 8 * half;
        if (m < N) out[(size_t)m * 64 + colB] = acc[r];
    }
}

// ---------------- seed: acc = xw * dinv^2 + bias (self-loop term) -----------
__global__ void k_seed(const float* __restrict__ xw, const float* __restrict__ dinv,
                       const float* __restrict__ bias, float* __restrict__ acc, int N) {
    int idx = blockIdx.x * blockDim.x + threadIdx.x;
    if (idx >= N * 64) return;
    int i = idx >> 6, f = idx & 63;
    float di = dinv[i];
    acc[idx] = xw[idx] * di * di + bias[f];
}

// ---------------- edge scatter: acc[dst] += xw[src] * dinv[s]*dinv[d] --------
// 16 threads per edge, float4 per thread: contiguous 256B gather per edge row.
__global__ __launch_bounds__(256) void k_edge(
    const long long* __restrict__ src, const long long* __restrict__ dst,
    const float* __restrict__ dinv, const float* __restrict__ xw,
    float* __restrict__ acc, int E) {
    int t = blockIdx.x * blockDim.x + threadIdx.x;
    int e = t >> 4;
    if (e >= E) return;
    int q = t & 15;
    int s = (int)src[e], d = (int)dst[e];
    float norm = dinv[s] * dinv[d];
    const float4 v = *(const float4*)(xw + (size_t)s * 64 + q * 4);
    float* p = acc + (size_t)d * 64 + q * 4;
    atomicAdd(p + 0, v.x * norm);
    atomicAdd(p + 1, v.y * norm);
    atomicAdd(p + 2, v.z * norm);
    atomicAdd(p + 3, v.w * norm);
}

// ---------------- launch ----------------
extern "C" void kernel_launch(void* const* d_in, const int* in_sizes, int n_in,
                              void* d_out, int out_size, void* d_ws, size_t ws_size,
                              hipStream_t stream) {
    const float*     x   = (const float*)d_in[0];
    const long long* ei  = (const long long*)d_in[1];
    const float*     W1  = (const float*)d_in[2];
    const float*     b1  = (const float*)d_in[3];
    const float*     W2  = (const float*)d_in[4];
    const float*     b2  = (const float*)d_in[5];
    float*           out = (float*)d_out;

    const int nhid  = in_sizes[3];               // 64
    const int nfeat = in_sizes[2] / nhid;        // 128
    const int N     = in_sizes[0] / nfeat;       // 50000
    const int E     = in_sizes[1] / 2;           // 800000
    (void)nhid; (void)out_size; (void)ws_size; (void)n_in;

    const long long* srcp = ei;
    const long long* dstp = ei + E;

    // workspace: dinv[N], xw[N*64], h1[N*64], hw[N*64]
    char* ws = (char*)d_ws;
    auto align256 = [](size_t v) { return (v + 255) & ~(size_t)255; };
    float* dinv = (float*)ws; ws += align256((size_t)N * sizeof(float));
    float* xw   = (float*)ws; ws += align256((size_t)N * 64 * sizeof(float));
    float* h1   = (float*)ws; ws += align256((size_t)N * 64 * sizeof(float));
    float* hw   = (float*)ws;

    const int T = 256;
    dim3 blk(T);

    // --- normalization ---
    k_deg_init<<<dim3((N + T - 1) / T), blk, 0, stream>>>(dinv, N);
    k_deg_count<<<dim3((E + T - 1) / T), blk, 0, stream>>>(dstp, dinv, E);
    k_rsqrt_inplace<<<dim3((N + T - 1) / T), blk, 0, stream>>>(dinv, N);

    // --- layer 1: xw = x @ W1; h1 = scatter(norm * xw) + b1 ---
    k_gemm_wmma<128, false><<<dim3((N + 31) / 32), blk, 0, stream>>>(x, W1, xw, N);
    k_seed<<<dim3((N * 64 + T - 1) / T), blk, 0, stream>>>(xw, dinv, b1, h1, N);
    k_edge<<<dim3((E * 16 + T - 1) / T), blk, 0, stream>>>(srcp, dstp, dinv, xw, h1, E);

    // --- layer 2: hw = leaky_relu(h1) @ W2 (ReLU fused into A-load); out = scatter + b2 ---
    k_gemm_wmma<64, true><<<dim3((N + 31) / 32), blk, 0, stream>>>(h1, W2, hw, N);
    k_seed<<<dim3((N * 64 + T - 1) / T), blk, 0, stream>>>(hw, dinv, b2, out, N);
    k_edge<<<dim3((E * 16 + T - 1) / T), blk, 0, stream>>>(srcp, dstp, dinv, hw, out, E);
}
